// GraphAttentionLayer_35081292873875
// MI455X (gfx1250) — matrix operand order repaired and checked by reading, not compiled, not run on hardware
//
#include <hip/hip_runtime.h>

typedef _Float16 half_t;
typedef __attribute__((ext_vector_type(16))) _Float16 v16h;
typedef __attribute__((ext_vector_type(8)))  float    v8f;
typedef __attribute__((ext_vector_type(4)))  int      v4i;
typedef __attribute__((ext_vector_type(4)))  float    v4f;

#define BATCH 4
#define NN    4096
#define CC    64
#define GAT_ALPHA 0.2f
#define LOG2E 1.4426950408889634f

__device__ __forceinline__ float lrelu(float x) { return fmaxf(x, GAT_ALPHA * x); }

// ---------------------------------------------------------------------------
// Kernel 1: h = x @ W (stored transposed as f16 hT[b][c][j]);
// f1 = (h.a1)*log2e, f2 = (h.a2)*log2e  (log2-domain softmax scores).
// One row (b,i) per 64-thread slice; 4 rows per 256-thread block.
// ---------------------------------------------------------------------------
__global__ __launch_bounds__(256) void gat_h_kernel(
    const float* __restrict__ x, const float* __restrict__ W,
    const float* __restrict__ a,
    half_t* __restrict__ hT, float* __restrict__ f1, float* __restrict__ f2)
{
    __shared__ float xs[4][CC];
    __shared__ float part1[8], part2[8];
    const int t    = threadIdx.x;
    const int lr   = t >> 6;          // row slice 0..3
    const int c    = t & 63;          // output channel
    const int rowg = blockIdx.x * 4 + lr;   // flat (b*N + i)
    const int b    = rowg >> 12;
    const int i    = rowg & (NN - 1);

    xs[lr][c] = x[(size_t)rowg * CC + c];
    __syncthreads();

    float acc = 0.0f;
#pragma unroll 8
    for (int k = 0; k < CC; ++k)
        acc = fmaf(xs[lr][k], W[k * CC + c], acc);

    hT[((size_t)(b * CC + c)) * NN + i] = (half_t)acc;

    float p1 = acc * a[c];
    float p2 = acc * a[CC + c];
#pragma unroll
    for (int off = 16; off > 0; off >>= 1) {
        p1 += __shfl_xor(p1, off);
        p2 += __shfl_xor(p2, off);
    }
    const int wave = t >> 5;
    if ((t & 31) == 0) { part1[wave] = p1; part2[wave] = p2; }
    __syncthreads();
    if (c == 0) {
        f1[rowg] = (part1[2 * lr] + part1[2 * lr + 1]) * LOG2E;
        f2[rowg] = (part2[2 * lr] + part2[2 * lr + 1]) * LOG2E;
    }
}

// ---------------------------------------------------------------------------
// Kernel 2: softmax max per row, all 4 batches per adj read (adj is shared).
// lrelu is monotonic and commutes with the positive log2e scaling, so
//   m[b][i] = lrelu(f1[b][i] + max_{j : adj[i][j]>0} f2[b][j])  (log2 domain)
// One wave per i; 8 waves per block.
// ---------------------------------------------------------------------------
__global__ __launch_bounds__(256) void gat_rowmax_kernel(
    const int* __restrict__ adj, const float* __restrict__ f1,
    const float* __restrict__ f2, float* __restrict__ m)
{
    const int wave = threadIdx.x >> 5;
    const int lane = threadIdx.x & 31;
    const int i    = blockIdx.x * 8 + wave;

    const int* arow = adj + (size_t)i * NN;

    float mx0 = -3.0e38f, mx1 = -3.0e38f, mx2 = -3.0e38f, mx3 = -3.0e38f;
    for (int j = lane; j < NN; j += 32) {
        if (arow[j] > 0) {
            mx0 = fmaxf(mx0, f2[0 * NN + j]);
            mx1 = fmaxf(mx1, f2[1 * NN + j]);
            mx2 = fmaxf(mx2, f2[2 * NN + j]);
            mx3 = fmaxf(mx3, f2[3 * NN + j]);
        }
    }
#pragma unroll
    for (int off = 16; off > 0; off >>= 1) {
        mx0 = fmaxf(mx0, __shfl_xor(mx0, off));
        mx1 = fmaxf(mx1, __shfl_xor(mx1, off));
        mx2 = fmaxf(mx2, __shfl_xor(mx2, off));
        mx3 = fmaxf(mx3, __shfl_xor(mx3, off));
    }
    if (lane == 0) {
        m[0 * NN + i] = (mx0 < -1.0e37f) ? 0.0f : lrelu(f1[0 * NN + i] + mx0);
        m[1 * NN + i] = (mx1 < -1.0e37f) ? 0.0f : lrelu(f1[1 * NN + i] + mx1);
        m[2 * NN + i] = (mx2 < -1.0e37f) ? 0.0f : lrelu(f1[2 * NN + i] + mx2);
        m[3 * NN + i] = (mx3 < -1.0e37f) ? 0.0f : lrelu(f1[3 * NN + i] + mx3);
    }
}

// ---------------------------------------------------------------------------
// Kernel 3: fused masked-softmax @ h via WMMA (flash-attention style).
// Block = 8 waves = one batch x 64 output rows. Wave (mt, jp): one 16-row
// M-tile, j-phase jp of 2 (j-range split, stride 64). Each k-step computes ONE
// A fragment (no redundancy) and multiplies it into ALL FOUR N-tiles (4 WMMA).
// Scalar (uniform) loop control keeps EXEC all-1s around the WMMAs. Scores are
// in the log2 domain: p = exp2(lrelu(f1+f2) - m) via a single v_exp_f32.
// Partial accumulators + row sums of the two j-phases merge via LDS once at
// the end; per-row softmax denominator distributed with ds_bpermute (__shfl).
// ---------------------------------------------------------------------------
__global__ __launch_bounds__(256) void gat_attn_kernel(
    const int* __restrict__ adj, const float* __restrict__ f1,
    const float* __restrict__ f2, const float* __restrict__ m,
    const half_t* __restrict__ hT, float* __restrict__ out)
{
    __shared__ float lds_acc[4][32][32];   // [mt][lane][4 frags x 8]
    __shared__ float lds_sum[4][32];       // [mt][lane]

    const int b    = blockIdx.x >> 6;
    const int ib   = blockIdx.x & 63;
    const int rowbase = ib * 64;
    const int tid  = threadIdx.x;
    const int wave = tid >> 5, lane = tid & 31;
    const int mt   = wave & 3,  jp  = wave >> 2;
    const int lid  = lane & 15, lh  = lane >> 4;

    // This lane's A-matrix row (fixed) and its softmax stats.
    const int   row = rowbase + mt * 16 + lid;
    const float f1r = f1[b * NN + row];
    const float mr  = m [b * NN + row];

    const int*    arow  = adj + (size_t)row * NN;
    const float*  f2b   = f2 + (size_t)b * NN;
    const half_t* hbase = hT + ((size_t)(b * CC)) * NN;
    const half_t* hb0 = hbase + (size_t)( 0 + lid) * NN;   // nt=0 channel
    const half_t* hb1 = hbase + (size_t)(16 + lid) * NN;   // nt=1
    const half_t* hb2 = hbase + (size_t)(32 + lid) * NN;   // nt=2
    const half_t* hb3 = hbase + (size_t)(48 + lid) * NN;   // nt=3

    v8f acc0 = {}, acc1 = {}, acc2 = {}, acc3 = {};
    float sp0 = 0.0f, sp1 = 0.0f, sp2 = 0.0f, sp3 = 0.0f;  // independent add chains

    // Uniform trip count -> scalar loop control (EXEC stays all-1s for WMMA).
    for (int it = 0; it < NN / 64; ++it) {
        const int jt = it * 64 + jp * 32;
        // A fragment K sets: lanes 0-15 -> K jt+0..15, lanes 16-31 -> K jt+16..31 (contiguous)
        const int base0 = jt + lh * 16;
        const int base1 = base0 + 8;
        v4i ad0 = *(const v4i*)(arow + base0);
        v4i ad1 = *(const v4i*)(arow + base0 + 4);
        v4i ad2 = *(const v4i*)(arow + base1);
        v4i ad3 = *(const v4i*)(arow + base1 + 4);
        v4f fv0 = *(const v4f*)(f2b + base0);
        v4f fv1 = *(const v4f*)(f2b + base0 + 4);
        v4f fv2 = *(const v4f*)(f2b + base1);
        v4f fv3 = *(const v4f*)(f2b + base1 + 4);

        v16h af;
#pragma unroll
        for (int e = 0; e < 4; ++e) {
            float s, p;
            s = f1r + fv0[e]; s = fmaxf(s, GAT_ALPHA * s);
            p = (ad0[e] > 0) ? __builtin_amdgcn_exp2f(s - mr) : 0.0f;
            sp0 += p; af[e]      = (half_t)p;
            s = f1r + fv1[e]; s = fmaxf(s, GAT_ALPHA * s);
            p = (ad1[e] > 0) ? __builtin_amdgcn_exp2f(s - mr) : 0.0f;
            sp1 += p; af[4 + e]  = (half_t)p;
            s = f1r + fv2[e]; s = fmaxf(s, GAT_ALPHA * s);
            p = (ad2[e] > 0) ? __builtin_amdgcn_exp2f(s - mr) : 0.0f;
            sp2 += p; af[8 + e]  = (half_t)p;
            s = f1r + fv3[e]; s = fmaxf(s, GAT_ALPHA * s);
            p = (ad3[e] > 0) ? __builtin_amdgcn_exp2f(s - mr) : 0.0f;
            sp3 += p; af[12 + e] = (half_t)p;
        }

        // B fragments: lanes 0-15 -> K jt+0..15, lanes 16-31 -> K jt+16..31 (contiguous)
        const int kb = jt + lh * 16;
        v16h bf0 = *(const v16h*)(hb0 + kb);
        v16h bf1 = *(const v16h*)(hb1 + kb);
        v16h bf2 = *(const v16h*)(hb2 + kb);
        v16h bf3 = *(const v16h*)(hb3 + kb);

        acc0 = __builtin_amdgcn_wmma_f32_16x16x32_f16(false, af, false, bf0,
                                                      (short)0, acc0, false, false);
        acc1 = __builtin_amdgcn_wmma_f32_16x16x32_f16(false, af, false, bf1,
                                                      (short)0, acc1, false, false);
        acc2 = __builtin_amdgcn_wmma_f32_16x16x32_f16(false, af, false, bf2,
                                                      (short)0, acc2, false, false);
        acc3 = __builtin_amdgcn_wmma_f32_16x16x32_f16(false, af, false, bf3,
                                                      (short)0, acc3, false, false);
    }

    // Row sum over this j-phase: this lane covered half the K range for its row;
    // the other half lives in lane^16.
    float sump = (sp0 + sp1) + (sp2 + sp3);
    float l = sump + __shfl_xor(sump, 16);

    // Merge the two j-phases through LDS.
    if (jp == 1) {
#pragma unroll
        for (int v = 0; v < 8; ++v) {
            lds_acc[mt][lane][v]      = acc0[v];
            lds_acc[mt][lane][8 + v]  = acc1[v];
            lds_acc[mt][lane][16 + v] = acc2[v];
            lds_acc[mt][lane][24 + v] = acc3[v];
        }
        lds_sum[mt][lane] = l;
    }
    __syncthreads();
    if (jp == 0) {
#pragma unroll
        for (int v = 0; v < 8; ++v) {
            acc0[v] += lds_acc[mt][lane][v];
            acc1[v] += lds_acc[mt][lane][8 + v];
            acc2[v] += lds_acc[mt][lane][16 + v];
            acc3[v] += lds_acc[mt][lane][24 + v];
        }
        l += lds_sum[mt][lane];

        // C layout: VGPR v holds row (v + 8*lh); lane lid holds column lid per N-tile.
        float* ob = out + ((size_t)(b * NN + rowbase + mt * 16)) * CC;
#pragma unroll
        for (int v = 0; v < 8; ++v) {
            const int rl = v + lh * 8;
            const float lr = __shfl(l, rl);      // lanes rl and rl+16 hold row rl's sum
            const float invl = 1.0f / fmaxf(lr, 1e-30f);
            ob[(size_t)rl * CC +  0 + lid] = fmaxf(acc0[v] * invl, 0.0f);
            ob[(size_t)rl * CC + 16 + lid] = fmaxf(acc1[v] * invl, 0.0f);
            ob[(size_t)rl * CC + 32 + lid] = fmaxf(acc2[v] * invl, 0.0f);
            ob[(size_t)rl * CC + 48 + lid] = fmaxf(acc3[v] * invl, 0.0f);
        }
    }
}

// ---------------------------------------------------------------------------
extern "C" void kernel_launch(void* const* d_in, const int* in_sizes, int n_in,
                              void* d_out, int out_size, void* d_ws, size_t ws_size,
                              hipStream_t stream) {
    const float* x   = (const float*)d_in[0];
    const int*   adj = (const int*)  d_in[1];
    const float* W   = (const float*)d_in[2];
    const float* a   = (const float*)d_in[3];
    float* out = (float*)d_out;

    char* ws = (char*)d_ws;
    half_t* hT = (half_t*)(ws);                                   // 2 MB
    float*  f1 = (float*)(ws + 2u * 1024 * 1024);                 // 64 KB
    float*  f2 = (float*)(ws + 2u * 1024 * 1024 + 64u * 1024);    // 64 KB
    float*  mm = (float*)(ws + 2u * 1024 * 1024 + 128u * 1024);   // 64 KB

    gat_h_kernel     <<<BATCH * NN / 4,  256, 0, stream>>>(x, W, a, hT, f1, f2);
    gat_rowmax_kernel<<<NN / 8,          256, 0, stream>>>(adj, f1, f2, mm);
    gat_attn_kernel  <<<BATCH * (NN/64), 256, 0, stream>>>(adj, f1, f2, mm, hT, out);
}
